// spatial_feature_modeling_layer_56891136803124
// MI455X (gfx1250) — matrix operand order repaired and checked by reading, not compiled, not run on hardware
//
#include <hip/hip_runtime.h>

// MI455X / gfx1250: wave32, WMMA bf16 16x16x32 with f32 accumulate.
typedef __attribute__((ext_vector_type(16))) __bf16 v16bf;
typedef __attribute__((ext_vector_type(8)))  __bf16 v8bf;
typedef __attribute__((ext_vector_type(8)))  float  v8f;

#define TPB 256

// ---------------------------------------------------------------- conversions
__global__ void cvt_f32_to_bf16(const float* __restrict__ in, __bf16* __restrict__ out, size_t n) {
  size_t i = (size_t)blockIdx.x * blockDim.x + threadIdx.x;
  if (i < n) out[i] = (__bf16)in[i];
}

// in: (R x C) f32 row-major  ->  out: (C x R) bf16 row-major
__global__ void transpose_f32_to_bf16(const float* __restrict__ in, __bf16* __restrict__ out,
                                      int R, int C) {
  size_t i = (size_t)blockIdx.x * blockDim.x + threadIdx.x;
  if (i >= (size_t)R * C) return;
  int r = (int)(i / C), c = (int)(i % C);
  out[(size_t)c * R + r] = (__bf16)in[i];
}

// invdeg[row] = 1 / max(sum_c adj[row][c], 1)
__global__ void invdeg_kernel(const float* __restrict__ adj, float* __restrict__ invdeg, int n) {
  __shared__ float red[TPB];
  int row = blockIdx.x;
  float s = 0.f;
  for (int c = threadIdx.x; c < n; c += TPB) s += adj[(size_t)row * n + c];
  red[threadIdx.x] = s;
  __syncthreads();
  for (int off = TPB / 2; off > 0; off >>= 1) {
    if ((int)threadIdx.x < off) red[threadIdx.x] += red[threadIdx.x + off];
    __syncthreads();
  }
  if (threadIdx.x == 0) invdeg[row] = 1.0f / fmaxf(red[0], 1.0f);
}

// ---------------------------------------------------------------- WMMA frags
// A 16x32 frag, K order chosen here (not the ISA table's): lanes 0-15 -> M=lane, K runs contiguous
// {klo..klo+15} with klo = (lane>=16)*16 (matches ldB). Two 16B loads.
// `p` is pre-offset to (row, klo); kb selects the K-step.
__device__ __forceinline__ v16bf ldA(const __bf16* __restrict__ p, int kb) {
  v8bf lo = *reinterpret_cast<const v8bf*>(p + kb);
  v8bf hi = *reinterpret_cast<const v8bf*>(p + kb + 8);
  v16bf r;
#pragma unroll
  for (int i = 0; i < 8; ++i) { r[i] = lo[i]; r[i + 8] = hi[i]; }
  return r;
}

// B 32x16 frag, K order chosen here (same as ldA): lane holds column n=lane&15, contiguous run of
// 16 K values starting at kb + (lane>=16)*16. Bt is B^T (N x K row-major), so
// this is one contiguous 32B run (two 16B loads). `p` pre-offset to (col, klo16).
__device__ __forceinline__ v16bf ldB(const __bf16* __restrict__ p, int kb) {
  v8bf lo = *reinterpret_cast<const v8bf*>(p + kb);
  v8bf hi = *reinterpret_cast<const v8bf*>(p + kb + 8);
  v16bf r;
#pragma unroll
  for (int i = 0; i < 8; ++i) { r[i] = lo[i]; r[i + 8] = hi[i]; }
  return r;
}

#define WMMA_BF16(a, b, c) \
  __builtin_amdgcn_wmma_f32_16x16x32_bf16(false, (a), false, (b), (short)0, (c), false, false)

// ---------------------------------------------------------------- GEMM
// C[z] (+)= A[z] (MxK row-major bf16) * Bt[z]^T  (Bt: NxK row-major bf16).
// Wave tile 32x64: 2 A-frags x 4 B-frags -> 8 WMMAs per K-step of 32, with
// explicit double-buffered prefetch of the next K-step's fragments so VMEM
// latency overlaps the WMMA burst. Block = 4 waves (2Mx2N) -> 64x128 region.
// grid = (M/64, N/128, nbatch).
__global__ void __launch_bounds__(128)
gemm_bf16_wmma(const __bf16* __restrict__ A, int lda, long long sA,
               const __bf16* __restrict__ Bt, int ldb, long long sB,
               float* __restrict__ C, int ldc, long long sC,
               int K, int accumulate) {
  const int lane = threadIdx.x & 31;
  const int wave = threadIdx.x >> 5;
  A  += (size_t)blockIdx.z * sA;
  Bt += (size_t)blockIdx.z * sB;
  C  += (size_t)blockIdx.z * sC;
  const int row0 = blockIdx.x * 64 + (wave & 1) * 32;
  const int col0 = blockIdx.y * 128 + (wave >> 1) * 64;
  const int m = lane & 15;

  // Pre-offset fragment base pointers (include lane-half K offsets).
  const __bf16* ap0 = A + (size_t)(row0 + m) * lda + ((lane >> 4) << 4);
  const __bf16* ap1 = ap0 + (size_t)16 * lda;
  const __bf16* bp0 = Bt + (size_t)(col0 +  0 + m) * ldb + ((lane >> 4) << 4);
  const __bf16* bp1 = Bt + (size_t)(col0 + 16 + m) * ldb + ((lane >> 4) << 4);
  const __bf16* bp2 = Bt + (size_t)(col0 + 32 + m) * ldb + ((lane >> 4) << 4);
  const __bf16* bp3 = Bt + (size_t)(col0 + 48 + m) * ldb + ((lane >> 4) << 4);

  v8f acc[2][4];
#pragma unroll
  for (int i = 0; i < 2; ++i)
#pragma unroll
    for (int j = 0; j < 4; ++j)
#pragma unroll
      for (int v = 0; v < 8; ++v) acc[i][j][v] = 0.0f;

  v16bf a0 = ldA(ap0, 0), a1 = ldA(ap1, 0);
  v16bf b0 = ldB(bp0, 0), b1 = ldB(bp1, 0), b2 = ldB(bp2, 0), b3 = ldB(bp3, 0);

  int kb = 0;
  for (; kb + 32 < K; kb += 32) {
    const int kn = kb + 32;
    // prefetch next K-step before issuing this step's WMMAs
    v16bf na0 = ldA(ap0, kn), na1 = ldA(ap1, kn);
    v16bf nb0 = ldB(bp0, kn), nb1 = ldB(bp1, kn);
    v16bf nb2 = ldB(bp2, kn), nb3 = ldB(bp3, kn);

    acc[0][0] = WMMA_BF16(a0, b0, acc[0][0]);
    acc[0][1] = WMMA_BF16(a0, b1, acc[0][1]);
    acc[0][2] = WMMA_BF16(a0, b2, acc[0][2]);
    acc[0][3] = WMMA_BF16(a0, b3, acc[0][3]);
    acc[1][0] = WMMA_BF16(a1, b0, acc[1][0]);
    acc[1][1] = WMMA_BF16(a1, b1, acc[1][1]);
    acc[1][2] = WMMA_BF16(a1, b2, acc[1][2]);
    acc[1][3] = WMMA_BF16(a1, b3, acc[1][3]);

    a0 = na0; a1 = na1; b0 = nb0; b1 = nb1; b2 = nb2; b3 = nb3;
  }
  // tail K-step
  acc[0][0] = WMMA_BF16(a0, b0, acc[0][0]);
  acc[0][1] = WMMA_BF16(a0, b1, acc[0][1]);
  acc[0][2] = WMMA_BF16(a0, b2, acc[0][2]);
  acc[0][3] = WMMA_BF16(a0, b3, acc[0][3]);
  acc[1][0] = WMMA_BF16(a1, b0, acc[1][0]);
  acc[1][1] = WMMA_BF16(a1, b1, acc[1][1]);
  acc[1][2] = WMMA_BF16(a1, b2, acc[1][2]);
  acc[1][3] = WMMA_BF16(a1, b3, acc[1][3]);

  // C/D layout: VGPR v -> M = v + (lane>=16 ? 8 : 0); N = lane & 15.
  const int rofs = (lane >> 4) * 8;
#pragma unroll
  for (int i = 0; i < 2; ++i) {
    const int rbase = row0 + i * 16 + rofs;
#pragma unroll
    for (int j = 0; j < 4; ++j) {
      const int col = col0 + j * 16 + m;
#pragma unroll
      for (int v = 0; v < 8; ++v) {
        float* cp = C + (size_t)(rbase + v) * ldc + col;
        if (accumulate) *cp += acc[i][j][v];
        else            *cp  = acc[i][j][v];
      }
    }
  }
}

// ---------------------------------------------------------------- epilogues
// f32 -> bf16 with optional bias(col), row-scale invdeg(row%N), relu;
// optionally also writes per-batch transpose outT[b][c][n] for the next adj@h.
__global__ void epilogue_to_bf16(const float* __restrict__ in,
                                 const float* __restrict__ bias,
                                 const float* __restrict__ invdeg,
                                 int relu,
                                 __bf16* __restrict__ out,
                                 __bf16* __restrict__ outT,
                                 int rowsPerBatch, int cols, long long total) {
  long long idx = (long long)blockIdx.x * blockDim.x + threadIdx.x;
  if (idx >= total) return;
  int c = (int)(idx % cols);
  long long r = idx / cols;
  int n = (int)(r % rowsPerBatch);
  int b = (int)(r / rowsPerBatch);
  float v = in[idx];
  if (bias)   v += bias[c];
  if (invdeg) v *= invdeg[n];
  if (relu)   v = fmaxf(v, 0.0f);
  __bf16 bv = (__bf16)v;
  out[idx] = bv;
  if (outT) outT[((size_t)b * cols + c) * rowsPerBatch + n] = bv;
}

__global__ void add_bias_f32(float* __restrict__ io, const float* __restrict__ bias,
                             int cols, long long total) {
  long long idx = (long long)blockIdx.x * blockDim.x + threadIdx.x;
  if (idx >= total) return;
  io[idx] += bias[(int)(idx % cols)];
}

// ---------------------------------------------------------------- launch
extern "C" void kernel_launch(void* const* d_in, const int* in_sizes, int n_in,
                              void* d_out, int out_size, void* d_ws, size_t ws_size,
                              hipStream_t stream) {
  (void)in_sizes; (void)n_in; (void)out_size; (void)ws_size;
  const int Bb = 4, Nn = 2048, Dd = 128, LD = 1536;   // L*D = 1536
  const int MN = Bb * Nn;                              // 8192 rows

  const float* x      = (const float*)d_in[0];
  const float* adj    = (const float*)d_in[1];
  const float* W_mlp2 = (const float*)d_in[2];
  const float* b_mlp2 = (const float*)d_in[3];
  const float* Wr[3]  = {(const float*)d_in[4], (const float*)d_in[7], (const float*)d_in[10]};
  const float* Wo[3]  = {(const float*)d_in[5], (const float*)d_in[8], (const float*)d_in[11]};
  const float* bL[3]  = {(const float*)d_in[6], (const float*)d_in[9], (const float*)d_in[12]};
  const float* W_mlp1 = (const float*)d_in[13];
  const float* b_mlp1 = (const float*)d_in[14];
  float* out = (float*)d_out;

  // ---- workspace carve (~47 MB total)
  size_t off = 0;
  auto carve = [&](size_t bytes) -> void* {
    void* p = (char*)d_ws + off;
    off += (bytes + 255) & ~(size_t)255;
    return p;
  };
  __bf16* adjb = (__bf16*)carve((size_t)Nn * Nn * 2);   // adj in bf16
  __bf16* xfb  = (__bf16*)carve((size_t)MN * LD * 2);   // flattened x in bf16
  __bf16* wm2t = (__bf16*)carve((size_t)Dd * LD * 2);   // W_mlp2^T (D x LD)
  __bf16* wm1t = (__bf16*)carve((size_t)LD * Dd * 2);   // W_mlp1^T (LD x D)
  __bf16 *wrt[3], *wot[3];
  for (int i = 0; i < 3; ++i) {
    wrt[i] = (__bf16*)carve((size_t)Dd * Dd * 2);
    wot[i] = (__bf16*)carve((size_t)Dd * Dd * 2);
  }
  __bf16* hcur = (__bf16*)carve((size_t)MN * Dd * 2);   // h (B*N x D)
  __bf16* hnxt = (__bf16*)carve((size_t)MN * Dd * 2);
  __bf16* hT   = (__bf16*)carve((size_t)MN * Dd * 2);   // h^T per batch (B x D x N)
  __bf16* aggb = (__bf16*)carve((size_t)MN * Dd * 2);
  float*  cbuf = (float*)carve((size_t)MN * Dd * 4);    // f32 accumulator buffer
  float*  idg  = (float*)carve((size_t)Nn * 4);

  auto nb = [](size_t n) { return (unsigned)((n + TPB - 1) / TPB); };
  dim3 gblk(128);
  const long long tot = (long long)MN * Dd;

  // ---- stage 0: convert / transpose / degrees
  cvt_f32_to_bf16<<<nb((size_t)MN * LD), TPB, 0, stream>>>(x, xfb, (size_t)MN * LD);
  cvt_f32_to_bf16<<<nb((size_t)Nn * Nn), TPB, 0, stream>>>(adj, adjb, (size_t)Nn * Nn);
  transpose_f32_to_bf16<<<nb((size_t)LD * Dd), TPB, 0, stream>>>(W_mlp2, wm2t, LD, Dd);
  transpose_f32_to_bf16<<<nb((size_t)Dd * LD), TPB, 0, stream>>>(W_mlp1, wm1t, Dd, LD);
  for (int i = 0; i < 3; ++i) {
    transpose_f32_to_bf16<<<nb((size_t)Dd * Dd), TPB, 0, stream>>>(Wr[i], wrt[i], Dd, Dd);
    transpose_f32_to_bf16<<<nb((size_t)Dd * Dd), TPB, 0, stream>>>(Wo[i], wot[i], Dd, Dd);
  }
  invdeg_kernel<<<Nn, TPB, 0, stream>>>(adj, idg, Nn);

  // ---- mlp2: h = xf @ W_mlp2 + b   (M=8192, N=128, K=1536)
  gemm_bf16_wmma<<<dim3(MN / 64, Dd / 128, 1), gblk, 0, stream>>>(
      xfb, LD, 0, wm2t, LD, 0, cbuf, Dd, 0, LD, 0);
  epilogue_to_bf16<<<nb((size_t)tot), TPB, 0, stream>>>(
      cbuf, b_mlp2, nullptr, 0, hcur, hT, Nn, Dd, tot);

  // ---- 3x DenseSAGEConv
  for (int i = 0; i < 3; ++i) {
    // agg = (adj @ h) / deg : per-batch M=2048, N=128, K=2048 (adj shared)
    gemm_bf16_wmma<<<dim3(Nn / 64, Dd / 128, Bb), gblk, 0, stream>>>(
        adjb, Nn, 0, hT, Nn, (long long)Dd * Nn, cbuf, Dd, (long long)Nn * Dd, Nn, 0);
    epilogue_to_bf16<<<nb((size_t)tot), TPB, 0, stream>>>(
        cbuf, nullptr, idg, 0, aggb, nullptr, Nn, Dd, tot);
    // h' = agg @ Wr + h @ Wo + b  (ReLU on first two layers)
    gemm_bf16_wmma<<<dim3(MN / 64, Dd / 128, 1), gblk, 0, stream>>>(
        aggb, Dd, 0, wrt[i], Dd, 0, cbuf, Dd, 0, Dd, 0);
    gemm_bf16_wmma<<<dim3(MN / 64, Dd / 128, 1), gblk, 0, stream>>>(
        hcur, Dd, 0, wot[i], Dd, 0, cbuf, Dd, 0, Dd, 1);
    epilogue_to_bf16<<<nb((size_t)tot), TPB, 0, stream>>>(
        cbuf, bL[i], nullptr, (i < 2) ? 1 : 0, hnxt, hT, Nn, Dd, tot);
    __bf16* t = hcur; hcur = hnxt; hnxt = t;
  }

  // ---- mlp1: out = h @ W_mlp1 + b  (M=8192, N=1536, K=128), f32 to d_out
  gemm_bf16_wmma<<<dim3(MN / 64, LD / 128, 1), gblk, 0, stream>>>(
      hcur, Dd, 0, wm1t, Dd, 0, out, LD, 0, Dd, 0);
  add_bias_f32<<<nb((size_t)MN * LD), TPB, 0, stream>>>(out, b_mlp1, LD, (long long)MN * LD);
}
